// GCN_64235530879310
// MI455X (gfx1250) — matrix operand order repaired and checked by reading, not compiled, not run on hardware
//
#include <hip/hip_runtime.h>
#include <math.h>

#define N_NODES 50000
#define N_EDGES 800000
#define F_IN    149
#define F_PAD   160      // 149 padded to 5*32 for the K dimension
#define H_DIM   256
#define N_GRAPHS 64
#define N_CLASSES 3
#define TPB 256

typedef __attribute__((ext_vector_type(16))) __bf16 v16bf;
typedef __attribute__((ext_vector_type(8)))  __bf16 v8bf;
typedef __attribute__((ext_vector_type(8)))  float  v8f;
typedef __attribute__((ext_vector_type(4)))  unsigned int v4u;
typedef __attribute__((ext_vector_type(8)))  int v8i;
typedef __attribute__((ext_vector_type(4)))  int v4i;

#if __has_builtin(__builtin_amdgcn_tensor_load_to_lds) && __has_builtin(__builtin_amdgcn_s_wait_tensorcnt)
#define USE_TDM 1
#else
#define USE_TDM 0
#endif

__device__ __forceinline__ unsigned short f32_to_bf16(float f) {
  unsigned int u = __float_as_uint(f);
  unsigned int r = u + 0x7FFFu + ((u >> 16) & 1u);   // round to nearest even
  return (unsigned short)(r >> 16);
}

#if USE_TDM
// Issue one TDM 2D tile load: global (rows x dwPerRow dwords, pitch dwPerRow) -> LDS.
// D# per ISA 8.3/8.4: group0 = {count=1 | lds_addr | global_addr(57b) | type=2},
// group1 = {data_size=4B, tensor_dim0/1, tile_dim0/1, dim0_stride}.
__device__ __forceinline__ void tdm_load_2d(const void* gptr, void* ldsptr,
                                            int dwPerRow, int rows) {
  unsigned long long ga = (unsigned long long)gptr;
  unsigned ldsOff = (unsigned)(unsigned long long)ldsptr;   // low 32b = LDS byte offset
  v4u g0 = { 1u,                                            // count=1, user desc
             ldsOff,
             (unsigned)ga,
             (unsigned)((ga >> 32) & 0x01FFFFFFu) | 0x80000000u };  // type=2
  v8i g1 = { 0x20000,                                       // data_size=2 -> 4 bytes
             dwPerRow << 16,                                // tensor_dim0[15:0]
             (rows & 0xFFFF) << 16,                         // tensor_dim1[15:0]
             dwPerRow << 16,                                // tile_dim0
             rows,                                          // tile_dim1 (tile_dim2=0)
             dwPerRow,                                      // tensor_dim0_stride[31:0]
             0, 0 };
  v4i z4 = {0, 0, 0, 0};
#if __clang_major__ >= 23
  v8i z8 = {0, 0, 0, 0, 0, 0, 0, 0};
  __builtin_amdgcn_tensor_load_to_lds(g0, g1, z4, z4, z8, 0);
#else
  __builtin_amdgcn_tensor_load_to_lds(g0, g1, z4, z4, 0);
#endif
}
#endif

// ---------------- utility fills / conversions ----------------

__global__ void k_fill_f32(float* p, float v, long long n) {
  long long i = (long long)blockIdx.x * blockDim.x + threadIdx.x;
  if (i < n) p[i] = v;
}

// convert float src[srcRows,srcCols] -> bf16 dst[dstRows,dstCols], zero padded
__global__ void k_pad_bf16(const float* __restrict__ src, unsigned short* __restrict__ dst,
                           int srcRows, int srcCols, int dstRows, int dstCols) {
  long long i = (long long)blockIdx.x * blockDim.x + threadIdx.x;
  long long n = (long long)dstRows * dstCols;
  if (i >= n) return;
  int r = (int)(i / dstCols), c = (int)(i % dstCols);
  float v = (r < srcRows && c < srcCols) ? src[(size_t)r * srcCols + c] : 0.0f;
  dst[i] = f32_to_bf16(v);
}

// ---------------- GCN normalization ----------------

__global__ void k_deg(const int* __restrict__ col, const float* __restrict__ w,
                      float* __restrict__ deg, int e_count) {
  int e = blockIdx.x * blockDim.x + threadIdx.x;
  if (e < e_count) atomicAdd(&deg[col[e]], w[e]);
}

__global__ void k_dinv(float* d, int n) {
  int i = blockIdx.x * blockDim.x + threadIdx.x;
  if (i >= n) return;
  float v = d[i];
  d[i] = (v > 0.0f) ? rsqrtf(v) : 0.0f;
}

__global__ void k_norm(const int* __restrict__ row, const int* __restrict__ col,
                       const float* __restrict__ w, const float* __restrict__ dinv,
                       float* __restrict__ norm, int e_count) {
  int e = blockIdx.x * blockDim.x + threadIdx.x;
  if (e < e_count) norm[e] = dinv[row[e]] * w[e] * dinv[col[e]];
}

// ---------------- WMMA GEMM: C[M,256] = A[M,K](bf16) * B[K,256](bf16) ----------------
// LDS-staged via the Tensor Data Mover: wave 0 issues two TENSOR_LOAD_TO_LDS DMAs
// (B panel [K,256] <=128KB, A panel [32,K] <=16KB), s_wait_tensorcnt 0, barrier;
// the k-loop then feeds WMMA purely from LDS (A: 16B loads, B: 2B gathers) (A/B read once from HBM per block).
// Fragment layouts per ISA 7.12.2:
//   A (16x32 bf16): lane -> M=lane%16; per lane two contiguous 16B chunks at
//                   k = kb + half*8 + [0..7] and kb + 16 + half*8 + [0..7], half=lane/16
//   B (32x16 bf16): lane -> N=lane%16; 16 K elements in the same order as A (strided LDS gather)
//   D (16x16 f32) : lane -> N=lane%16; VGPR r -> M = r + 8*(lane/16)
__global__ __launch_bounds__(TPB)
void k_gemm_bf16_n256(const unsigned short* __restrict__ A,
                      const unsigned short* __restrict__ B,
                      float* __restrict__ C, int M, int K) {
  extern __shared__ __align__(64) unsigned short lds[];
  unsigned short* Bs = lds;                       // [K][256]
  unsigned short* As = lds + (size_t)K * H_DIM;   // [32][K]

  int tid = threadIdx.x;
  int lane = tid & 31;
  int w    = tid >> 5;
  int mBlock = blockIdx.x * 32;
  int rows = M - mBlock; if (rows > 32) rows = 32;

#if USE_TDM
  if (w == 0) {                                   // one wave drives both DMAs
    tdm_load_2d(B, Bs, H_DIM / 2, K);             // B panel: K rows of 128 dwords
    tdm_load_2d(A + (size_t)mBlock * K, As, K / 2, rows);
    __builtin_amdgcn_s_wait_tensorcnt(0);
  }
#else
  {
    const uint4* src = (const uint4*)B;
    uint4* dst = (uint4*)Bs;
    int nchunk = K * (H_DIM / 8);
    for (int i = tid; i < nchunk; i += TPB) dst[i] = src[i];
  }
  {
    const uint4* src = (const uint4*)(A + (size_t)mBlock * K);
    uint4* dst = (uint4*)As;
    int nchunk = rows * (K / 8);
    for (int i = tid; i < nchunk; i += TPB) dst[i] = src[i];
  }
#endif
  __syncthreads();

  int mLocal = (w >> 2) * 16;                     // 0 or 16
  int nBase  = (w & 3) * 64;
  if (mBlock + mLocal >= M) return;               // wave-uniform exit (after barrier)

  int half = lane >> 4;
  const unsigned short* Arow = As + (size_t)(mLocal + (lane & 15)) * K + half * 8;
  const unsigned short* Bcol = Bs + nBase + (lane & 15);

  v8f zero = {0.f,0.f,0.f,0.f,0.f,0.f,0.f,0.f};
  v8f acc[4]; acc[0]=zero; acc[1]=zero; acc[2]=zero; acc[3]=zero;

  for (int kb = 0; kb < K; kb += 32) {
    v8bf alo = *(const v8bf*)(Arow + kb);
    v8bf ahi = *(const v8bf*)(Arow + kb + 16);
    v16bf av = __builtin_shufflevector(alo, ahi, 0,1,2,3,4,5,6,7,8,9,10,11,12,13,14,15);
    const unsigned short* Brow = Bcol + (size_t)kb * H_DIM;
    // load all four B fragments first, then issue four back-to-back WMMAs
    v16bf bv0, bv1, bv2, bv3;
#pragma unroll
    for (int s = 0; s < 16; ++s) {
      const unsigned short* bp = Brow + (size_t)((s < 8 ? s : s + 8) + half * 8) * H_DIM;
      bv0[s] = __builtin_bit_cast(__bf16, bp[0]);
      bv1[s] = __builtin_bit_cast(__bf16, bp[16]);
      bv2[s] = __builtin_bit_cast(__bf16, bp[32]);
      bv3[s] = __builtin_bit_cast(__bf16, bp[48]);
    }
    acc[0] = __builtin_amdgcn_wmma_f32_16x16x32_bf16(false, av, false, bv0, (short)0, acc[0], false, false);
    acc[1] = __builtin_amdgcn_wmma_f32_16x16x32_bf16(false, av, false, bv1, (short)0, acc[1], false, false);
    acc[2] = __builtin_amdgcn_wmma_f32_16x16x32_bf16(false, av, false, bv2, (short)0, acc[2], false, false);
    acc[3] = __builtin_amdgcn_wmma_f32_16x16x32_bf16(false, av, false, bv3, (short)0, acc[3], false, false);
  }

  float* Crow = C + (size_t)(mBlock + mLocal + half * 8) * H_DIM + nBase + (lane & 15);
#pragma unroll
  for (int t = 0; t < 4; ++t)
#pragma unroll
    for (int r = 0; r < 8; ++r)
      Crow[(size_t)r * H_DIM + t * 16] = acc[t][r];
}

// ---------------- edge gather / scatter-add (L2-resident) ----------------
// One wave32 per edge; lane handles 8 consecutive features (two float4 loads, 8 f32 atomics).
__global__ void k_scatter(const float* __restrict__ h, const float* __restrict__ norm,
                          const int* __restrict__ row, const int* __restrict__ col,
                          float* __restrict__ out, int e_count) {
  long long gt = (long long)blockIdx.x * blockDim.x + threadIdx.x;
  int e = (int)(gt >> 5);
  if (e >= e_count) return;
  int l = (int)(gt & 31);
  float nv = norm[e];
  const float4* hp = (const float4*)(h + (size_t)row[e] * H_DIM) + l * 2;
  float4 a = hp[0], b = hp[1];
  float* op = out + (size_t)col[e] * H_DIM + l * 8;
  atomicAdd(op + 0, a.x * nv); atomicAdd(op + 1, a.y * nv);
  atomicAdd(op + 2, a.z * nv); atomicAdd(op + 3, a.w * nv);
  atomicAdd(op + 4, b.x * nv); atomicAdd(op + 5, b.y * nv);
  atomicAdd(op + 6, b.z * nv); atomicAdd(op + 7, b.w * nv);
}

__global__ void k_bias_act(float* __restrict__ h, const float* __restrict__ b,
                           long long n, int relu) {
  long long i = (long long)blockIdx.x * blockDim.x + threadIdx.x;
  if (i >= n) return;
  float v = h[i] + b[(int)(i & (H_DIM - 1))];
  if (relu) v = v > 0.0f ? v : 0.0f;
  h[i] = v;
}

// ---------------- attention pooling ----------------

__global__ void k_gate(const float* __restrict__ h, const float* __restrict__ gw,
                       const float* __restrict__ gb, float* __restrict__ g, int n) {
  int i = blockIdx.x * blockDim.x + threadIdx.x;
  if (i >= n) return;
  const float* hr = h + (size_t)i * H_DIM;
  float s = 0.0f;
  for (int k = 0; k < H_DIM; ++k) s += hr[k] * gw[k];
  g[i] = s + gb[0];
}

__device__ __forceinline__ void atomicMaxF(float* a, float v) {
  if (v >= 0.0f) atomicMax((int*)a, __float_as_int(v));
  else           atomicMin((unsigned int*)a, __float_as_uint(v));
}

__global__ void k_segmax(const float* __restrict__ g, const int* __restrict__ batch,
                         float* __restrict__ m, int n) {
  int i = blockIdx.x * blockDim.x + threadIdx.x;
  if (i < n) atomicMaxF(&m[batch[i]], g[i]);
}

__global__ void k_expsum(float* __restrict__ g, const int* __restrict__ batch,
                         const float* __restrict__ m, float* __restrict__ s, int n) {
  int i = blockIdx.x * blockDim.x + threadIdx.x;
  if (i >= n) return;
  float e = __expf(g[i] - m[batch[i]]);
  g[i] = e;
  atomicAdd(&s[batch[i]], e);
}

__global__ void k_pool(const float* __restrict__ h, const float* __restrict__ g,
                       const float* __restrict__ s, const int* __restrict__ batch,
                       float* __restrict__ pooled, long long n) {
  long long i = (long long)blockIdx.x * blockDim.x + threadIdx.x;
  if (i >= n) return;
  int node = (int)(i >> 8), c = (int)(i & 255);
  int b = batch[node];
  float gate = g[node] / (s[b] + 1e-16f);
  atomicAdd(&pooled[(size_t)b * H_DIM + c], gate * h[i]);
}

// ---------------- final MLP (tiny) ----------------

__global__ void k_mlp1(const float* __restrict__ pooled, const float* __restrict__ W,
                       const float* __restrict__ b, float* __restrict__ t1) {
  int i = blockIdx.x * blockDim.x + threadIdx.x;
  if (i >= N_GRAPHS * H_DIM) return;
  int gi = i / H_DIM, c = i % H_DIM;
  const float* pr = pooled + (size_t)gi * H_DIM;
  float s = 0.0f;
  for (int k = 0; k < H_DIM; ++k) s += pr[k] * W[(size_t)k * H_DIM + c];
  s += b[c];
  t1[i] = s > 0.0f ? s : 0.0f;
}

__global__ void k_mlp2(const float* __restrict__ t1, const float* __restrict__ W,
                       const float* __restrict__ b, float* __restrict__ out) {
  int i = blockIdx.x * blockDim.x + threadIdx.x;
  if (i >= N_GRAPHS * N_CLASSES) return;
  int gi = i / N_CLASSES, c = i % N_CLASSES;
  const float* tr = t1 + (size_t)gi * H_DIM;
  float s = 0.0f;
  for (int k = 0; k < H_DIM; ++k) s += tr[k] * W[k * N_CLASSES + c];
  out[i] = s + b[c];
}

// ---------------- host orchestration ----------------

extern "C" void kernel_launch(void* const* d_in, const int* in_sizes, int n_in,
                              void* d_out, int out_size, void* d_ws, size_t ws_size,
                              hipStream_t stream) {
  (void)in_sizes; (void)n_in; (void)out_size; (void)ws_size;
  const float* x   = (const float*)d_in[0];
  const int*   ei  = (const int*)d_in[1];
  const float* ew  = (const float*)d_in[2];
  const int*   bat = (const int*)d_in[3];
  const float* W1 = (const float*)d_in[4];  const float* b1 = (const float*)d_in[5];
  const float* W2 = (const float*)d_in[6];  const float* b2 = (const float*)d_in[7];
  const float* W3 = (const float*)d_in[8];  const float* b3 = (const float*)d_in[9];
  const float* gw = (const float*)d_in[10]; const float* gb = (const float*)d_in[11];
  const float* l1w = (const float*)d_in[12]; const float* l1b = (const float*)d_in[13];
  const float* l2w = (const float*)d_in[14]; const float* l2b = (const float*)d_in[15];
  float* out = (float*)d_out;
  const int* row = ei;
  const int* col = ei + N_EDGES;

  // workspace carve-out (~132 MB)
  char* p = (char*)d_ws;
  auto take = [&](size_t bytes) -> char* {
    char* r = p; p += (bytes + 255) & ~(size_t)255; return r;
  };
  float* norm   = (float*)take(sizeof(float) * N_EDGES);
  float* dinv   = (float*)take(sizeof(float) * N_NODES);
  float* gbuf   = (float*)take(sizeof(float) * N_NODES);
  float* mbuf   = (float*)take(sizeof(float) * N_GRAPHS);
  float* sbuf   = (float*)take(sizeof(float) * N_GRAPHS);
  float* pooled = (float*)take(sizeof(float) * N_GRAPHS * H_DIM);
  float* t1     = (float*)take(sizeof(float) * N_GRAPHS * H_DIM);
  unsigned short* Abf = (unsigned short*)take(sizeof(unsigned short) * (size_t)N_NODES * H_DIM);
  unsigned short* Wbf = (unsigned short*)take(sizeof(unsigned short) * H_DIM * H_DIM);
  float* hg = (float*)take(sizeof(float) * (size_t)N_NODES * H_DIM);
  float* ha = (float*)take(sizeof(float) * (size_t)N_NODES * H_DIM);

  const int T = TPB;
  auto g1 = [&](long long n) { return dim3((unsigned)((n + T - 1) / T)); };
  const long long NH = (long long)N_NODES * H_DIM;

  // edge normalization coefficients (computed once, reused for all 3 layers)
  k_fill_f32<<<g1(N_NODES), T, 0, stream>>>(dinv, 0.0f, N_NODES);
  k_deg <<<g1(N_EDGES), T, 0, stream>>>(col, ew, dinv, N_EDGES);
  k_dinv<<<g1(N_NODES), T, 0, stream>>>(dinv, N_NODES);
  k_norm<<<g1(N_EDGES), T, 0, stream>>>(row, col, ew, dinv, norm, N_EDGES);

  const float* Ws[3] = {W1, W2, W3};
  const float* bs[3] = {b1, b2, b3};
  for (int l = 0; l < 3; ++l) {
    int K = (l == 0) ? F_PAD : H_DIM;
    if (l == 0) {
      k_pad_bf16<<<g1((long long)N_NODES * F_PAD), T, 0, stream>>>(x, Abf, N_NODES, F_IN, N_NODES, F_PAD);
      k_pad_bf16<<<g1((long long)F_PAD * H_DIM), T, 0, stream>>>(Ws[0], Wbf, F_IN, H_DIM, F_PAD, H_DIM);
    } else {
      k_pad_bf16<<<g1(NH), T, 0, stream>>>(ha, Abf, N_NODES, H_DIM, N_NODES, H_DIM);
      k_pad_bf16<<<g1((long long)H_DIM * H_DIM), T, 0, stream>>>(Ws[l], Wbf, H_DIM, H_DIM, H_DIM, H_DIM);
    }
    // LDS: B panel K*256 bf16 + A panel 32*K bf16 = K*576 bytes (144KB max)
    size_t smem = (size_t)K * (H_DIM * 2 + 64);
    int blocksM = (N_NODES + 31) / 32;
    k_gemm_bf16_n256<<<dim3(blocksM), T, smem, stream>>>(Abf, Wbf, hg, N_NODES, K);
    k_fill_f32<<<g1(NH), T, 0, stream>>>(ha, 0.0f, NH);
    k_scatter<<<g1((long long)N_EDGES * 32), T, 0, stream>>>(hg, norm, row, col, ha, N_EDGES);
    k_bias_act<<<g1(NH), T, 0, stream>>>(ha, bs[l], NH, (l < 2) ? 1 : 0);
  }

  // attention pooling
  k_gate<<<g1(N_NODES), T, 0, stream>>>(ha, gw, gb, gbuf, N_NODES);
  k_fill_f32<<<1, T, 0, stream>>>(mbuf, -3.402823466e38f, N_GRAPHS);
  k_fill_f32<<<1, T, 0, stream>>>(sbuf, 0.0f, N_GRAPHS);
  k_segmax<<<g1(N_NODES), T, 0, stream>>>(gbuf, bat, mbuf, N_NODES);
  k_expsum<<<g1(N_NODES), T, 0, stream>>>(gbuf, bat, mbuf, sbuf, N_NODES);
  k_fill_f32<<<g1(N_GRAPHS * H_DIM), T, 0, stream>>>(pooled, 0.0f, N_GRAPHS * H_DIM);
  k_pool<<<g1(NH), T, 0, stream>>>(ha, gbuf, sbuf, bat, pooled, NH);

  // final MLP
  k_mlp1<<<g1(N_GRAPHS * H_DIM), T, 0, stream>>>(pooled, l1w, l1b, t1);
  k_mlp2<<<g1(N_GRAPHS * N_CLASSES), T, 0, stream>>>(t1, l2w, l2b, out);
}
